// RNNBase_25950192402935
// MI455X (gfx1250) — matrix operand order repaired and checked by reading, not compiled, not run on hardware
//
#include <hip/hip_runtime.h>

// ---------------------------------------------------------------------------
// 2-layer LSTM, T=512, B=64, H=1024, bf16 WMMA (v_wmma_f32_16x16x32_bf16),
// f32 accumulation + f32 cell state. One fused kernel per timestep per layer.
// Per tile: a PAIR of waves split K (x-part / h-part), software-pipelined
// fragment loads, LDS reduction, in-register gate math.
// ---------------------------------------------------------------------------

#define T_STEPS 512
#define BATCH   64
#define HID     1024
#define NLAYER  2
#define GATES   (4 * HID)   // 4096

typedef __bf16 bf16;
typedef __bf16 v16bf __attribute__((ext_vector_type(16)));
typedef __bf16 v8bf  __attribute__((ext_vector_type(8)));
typedef float  v8f   __attribute__((ext_vector_type(8)));

// ---------------- prep kernels ----------------

__global__ void cvt_f32_bf16(const float* __restrict__ in, bf16* __restrict__ out, int n) {
    int i = blockIdx.x * blockDim.x + threadIdx.x;
    int stride = gridDim.x * blockDim.x;
    for (; i < n; i += stride) out[i] = (bf16)in[i];
}

__global__ void bias_sum_k(const float* __restrict__ a, const float* __restrict__ b,
                           float* __restrict__ o, int n) {
    int i = blockIdx.x * blockDim.x + threadIdx.x;
    if (i < n) o[i] = a[i] + b[i];
}

__global__ void zero_f32(float* __restrict__ p, int n) {
    int i = blockIdx.x * blockDim.x + threadIdx.x;
    int stride = gridDim.x * blockDim.x;
    for (; i < n; i += stride) p[i] = 0.0f;
}

// ---------------- fragment structures ----------------

struct Frag {
    v16bf a;        // A 16x32 bf16 fragment
    v16bf b[4];     // B 32x16 bf16 fragments, one per gate
};

// A 16x32 bf16 layout: lane<16 holds row lane, K kk+0..15 contiguous;
//                      lane>=16 holds row lane-16, K kk+16..31 contiguous.
// B 32x16 bf16 layout: lane<16 holds col lane, K kk+0..15 contiguous;
//                      lane>=16 holds col lane-16, K kk+16..31 contiguous.
__device__ __forceinline__ void load_chunk(Frag& f,
                                           const bf16* __restrict__ Arow, int kAoff,
                                           const bf16* const* Bp, int kk) {
    v8bf a0 = *(const v8bf*)(Arow + kk + kAoff);
    v8bf a1 = *(const v8bf*)(Arow + kk + kAoff + 8);
    union { v16bf v; v8bf h[2]; } u;
    u.h[0] = a0; u.h[1] = a1;
    f.a = u.v;
#pragma unroll
    for (int g = 0; g < 4; ++g)
        f.b[g] = *(const v16bf*)(Bp[g] + kk);
}

// ---------------- fused LSTM timestep ----------------
// Grid: 64 blocks x 256 threads = 512 waves = 256 tiles x 2 partner waves.
// Partner 0: x_t @ Wx^T (K=1024); partner 1: h_{t-1} @ Wh^T (K=1024).
// LDS reduce, then partner 0 applies gates + state update.

__global__ __launch_bounds__(256) void lstm_step(
    const bf16*  __restrict__ X,      // [B, H] bf16 input at time t
    const bf16*  __restrict__ Wx,     // [4H, H] bf16
    const bf16*  __restrict__ Wh,     // [4H, H] bf16
    const float* __restrict__ bsum,   // [4H] = bxh + bhh
    const bf16*  __restrict__ hin,    // [B, H] bf16
    bf16*        __restrict__ hout,   // [B, H] bf16 (ping-pong)
    float*       __restrict__ cstate, // [B, H] f32 (exclusively owned tiles)
    bf16*        __restrict__ ybf,    // [B, H] bf16 inter-layer bounce (layer 0)
    float*       __restrict__ yf32,   // [B, H] f32 slice of output (layer 1)
    float*       __restrict__ hid_fin,// [B, H] final h (written at t == T-1)
    float*       __restrict__ cel_fin,// [B, H] final c
    int is_last_t, int store_f32)
{
    __shared__ float red[4][4][8][32];   // [pair][gate][r][lane] = 16 KB

    const int lane = threadIdx.x & 31;
    const int wave = threadIdx.x >> 5;
    const int pair = wave >> 1;          // 0..3: tile slot within block
    const int part = wave & 1;           // 0: x-part, 1: h-part (wave-uniform)
    const int tile = blockIdx.x * 4 + pair;  // 0..255
    const int btile = tile >> 6;         // 0..3   (batch / 16)
    const int jtile = tile & 63;         // 0..63  (hid / 16)

    const int rowA  = (btile << 4) + (lane & 15);
    const int kAoff = (lane >> 4) << 4;  // 0 or 16
    const int nB    = lane & 15;
    const int kBoff = (lane >> 4) << 4;  // 0 or 16

    const bf16* Arow  = (part ? hin : X) + (size_t)rowA * HID;
    const bf16* Wbase = part ? Wh : Wx;

    const bf16* Bp[4];
#pragma unroll
    for (int g = 0; g < 4; ++g)
        Bp[g] = Wbase + (size_t)(g * HID + (jtile << 4) + nB) * HID + kBoff;

    v8f acc[4];
#pragma unroll
    for (int g = 0; g < 4; ++g) acc[g] = (v8f)(0.0f);

    // ---- software-pipelined K loop: 32 chunks of 32 ----
    Frag fr[2];
    load_chunk(fr[0], Arow, kAoff, Bp, 0);
#pragma unroll 2
    for (int ck = 0; ck < 32; ++ck) {
        const int cur = ck & 1;
        const int nxt = cur ^ 1;
        if (ck < 31)
            load_chunk(fr[nxt], Arow, kAoff, Bp, (ck + 1) << 5);
#pragma unroll
        for (int g = 0; g < 4; ++g)
            acc[g] = __builtin_amdgcn_wmma_f32_16x16x32_bf16(
                false, fr[cur].a, false, fr[cur].b[g], (short)0, acc[g], false, false);
    }

    // ---- LDS reduction across the partner pair ----
    if (part) {
#pragma unroll
        for (int g = 0; g < 4; ++g)
#pragma unroll
            for (int r = 0; r < 8; ++r)
                red[pair][g][r][lane] = acc[g][r];
    }
    __syncthreads();
    if (part) return;

#pragma unroll
    for (int g = 0; g < 4; ++g)
#pragma unroll
        for (int r = 0; r < 8; ++r)
            acc[g][r] += red[pair][g][r][lane];

    // ---- gates + state update, in-register ----
    // C/D f32 layout: VGPR r, lane<16 -> (M=r, N=lane); lane>=16 -> (M=8+r, N=lane-16)
    const int j  = (jtile << 4) + nB;
    const float bi = bsum[j];
    const float bf = bsum[HID + j];
    const float bc = bsum[2 * HID + j];
    const float bo = bsum[3 * HID + j];
    const int m0 = (lane >> 4) << 3;

#pragma unroll
    for (int r = 0; r < 8; ++r) {
        const int b   = (btile << 4) + m0 + r;
        const int idx = b * HID + j;
        float gi = acc[0][r] + bi;
        float gf = acc[1][r] + bf;
        float gc = acc[2][r] + bc;
        float go = acc[3][r] + bo;
        float i_ = 1.0f / (1.0f + __expf(-gi));
        float f_ = 1.0f / (1.0f + __expf(-gf));
        float cg = tanhf(gc);
        float o_ = 1.0f / (1.0f + __expf(-go));
        float cn = f_ * cstate[idx] + i_ * cg;
        float hn = o_ * tanhf(cn);
        cstate[idx] = cn;
        hout[idx]   = (bf16)hn;
        if (store_f32) yf32[idx] = hn;
        else           ybf[idx]  = (bf16)hn;
        if (is_last_t) { hid_fin[idx] = hn; cel_fin[idx] = cn; }
    }
}

// ---------------- host orchestration ----------------

extern "C" void kernel_launch(void* const* d_in, const int* in_sizes, int n_in,
                              void* d_out, int out_size, void* d_ws, size_t ws_size,
                              hipStream_t stream) {
    (void)in_sizes; (void)n_in; (void)out_size; (void)ws_size;

    const float* x    = (const float*)d_in[0];  // [T, B, H]
    const float* Wxh  = (const float*)d_in[1];  // [L, 4H, H]
    const float* bxh  = (const float*)d_in[2];  // [L, 4H]
    const float* Whh  = (const float*)d_in[3];  // [L, 4H, H]
    const float* bhh  = (const float*)d_in[4];  // [L, 4H]
    float* out = (float*)d_out;                 // output | hidden | cell

    char* ws = (char*)d_ws;
    const size_t szW  = (size_t)NLAYER * GATES * HID * sizeof(bf16);   // 16 MB
    const size_t szX  = (size_t)T_STEPS * BATCH * HID * sizeof(bf16);  // 64 MB
    const size_t szBH16 = (size_t)BATCH * HID * sizeof(bf16);          // 128 KB
    const size_t szBH32 = (size_t)BATCH * HID * sizeof(float);         // 256 KB

    bf16*  Wxbf = (bf16*)(ws);
    bf16*  Whbf = (bf16*)(ws + szW);
    float* bs   = (float*)(ws + 2 * szW);
    bf16*  xbf  = (bf16*)(ws + 2 * szW + (size_t)NLAYER * GATES * sizeof(float));
    bf16*  ybf  = (bf16*)((char*)xbf + szX);
    bf16*  hbuf = (bf16*)((char*)ybf + szBH16);      // 4 x [B,H] bf16: layer0 pp0/pp1, layer1 pp0/pp1
    float* cbuf = (float*)((char*)hbuf + 4 * szBH16);// 2 x [B,H] f32

    const int nW = NLAYER * GATES * HID;        // 8388608
    const int nX = T_STEPS * BATCH * HID;       // 33554432
    const int nBH = BATCH * HID;                // 65536

    cvt_f32_bf16<<<2048, 256, 0, stream>>>(Wxh, Wxbf, nW);
    cvt_f32_bf16<<<2048, 256, 0, stream>>>(Whh, Whbf, nW);
    bias_sum_k <<<(NLAYER * GATES + 255) / 256, 256, 0, stream>>>(bxh, bhh, bs, NLAYER * GATES);
    cvt_f32_bf16<<<4096, 256, 0, stream>>>(x, xbf, nX);
    // zero h ping-pong (bf16, as 32-bit words) + c state (f32): 1 MB total
    zero_f32<<<512, 256, 0, stream>>>((float*)hbuf, (int)((4 * szBH16 + 2 * szBH32) / 4));

    float* hid = out + (size_t)T_STEPS * BATCH * HID;
    float* cel = hid + (size_t)NLAYER * BATCH * HID;

    for (int t = 0; t < T_STEPS; ++t) {
        const int pp = t & 1;
        const int last = (t == T_STEPS - 1);
        // layer 0: reads xbf[t], writes ybf (bf16 bounce)
        lstm_step<<<64, 256, 0, stream>>>(
            xbf + (size_t)t * nBH,
            Wxbf, Whbf, bs,
            hbuf + (size_t)pp * nBH, hbuf + (size_t)(1 - pp) * nBH,
            cbuf,
            ybf, /*yf32=*/nullptr,
            hid, cel,
            last, /*store_f32=*/0);
        // layer 1: reads ybf, writes f32 output slice
        lstm_step<<<64, 256, 0, stream>>>(
            ybf,
            Wxbf + (size_t)GATES * HID, Whbf + (size_t)GATES * HID, bs + GATES,
            hbuf + (size_t)(2 + pp) * nBH, hbuf + (size_t)(2 + (1 - pp)) * nBH,
            cbuf + nBH,
            /*ybf=*/nullptr, out + (size_t)t * nBH,
            hid + nBH, cel + nBH,
            last, /*store_f32=*/1);
    }
}
